// DecoderCell_52793738003119
// MI455X (gfx1250) — matrix operand order repaired and checked by reading, not compiled, not run on hardware
//
#include <hip/hip_runtime.h>
#include <cstdint>
#include <cstddef>

#define B_SZ 512
#define T_SZ 512
#define DECD 512
#define ATTD 512
#define LISD 256
#define VOCD 123
#define DIN1 (DECD + VOCD + 2 * LISD) /* 1147 */
#define RNNI_LD 1152                  /* DIN1 padded to multiple of 32 */
#define LDSP 40                       /* LDS row stride in bf16 (80B, 16B-aligned) */

typedef __attribute__((ext_vector_type(16))) __bf16 v16bf;
typedef __attribute__((ext_vector_type(8)))  float  v8f;
typedef __attribute__((ext_vector_type(4)))  float  f32x4;

union Frag { v16bf v; int4 q[2]; };

// ---------------------------------------------------------------------------
// Threefry-2x32 (16 rounds), matching JAX's PRNG structure
// ---------------------------------------------------------------------------
__device__ __forceinline__ void tf2x32(uint32_t k0, uint32_t k1,
                                       uint32_t x0, uint32_t x1,
                                       uint32_t& o0, uint32_t& o1) {
  uint32_t ks2 = 0x1BD11BDAu ^ k0 ^ k1;
#define TFR(r) { x0 += x1; x1 = (x1 << (r)) | (x1 >> (32 - (r))); x1 ^= x0; }
  x0 += k0; x1 += k1;
  TFR(13) TFR(15) TFR(26) TFR(6)
  x0 += k1; x1 += ks2 + 1u;
  TFR(17) TFR(29) TFR(16) TFR(24)
  x0 += ks2; x1 += k0 + 2u;
  TFR(13) TFR(15) TFR(26) TFR(6)
  x0 += k0; x1 += k1 + 3u;
  TFR(17) TFR(29) TFR(16) TFR(24)
  x0 += k1; x1 += ks2 + 4u;
#undef TFR
  o0 = x0; o1 = x1;
}

__device__ __forceinline__ float bits_to_u01(uint32_t bits) {
  return __uint_as_float((bits >> 9) | 0x3f800000u) - 1.0f;
}

// ---------------------------------------------------------------------------
// Kernel 1: scheduled-sampling blend + concat([psv, ytu, pcv]) -> rnni (ld 1152)
// ---------------------------------------------------------------------------
__global__ __launch_bounds__(128)
void sample_concat_kernel(const float* __restrict__ yin, const float* __restrict__ pyp,
                          const float* __restrict__ psv, const float* __restrict__ pcv,
                          const int* __restrict__ blend, float* __restrict__ rnni) {
  const int b = blockIdx.x, tid = threadIdx.x;
  uint32_t r02x, r02y, r13x, r13y;
  tf2x32(0u, 1u, 0u, 2u, r02x, r02y);
  tf2x32(0u, 1u, 1u, 3u, r13x, r13y);
  const uint32_t k1a = r02x, k1b = r13x; // categorical key
  const uint32_t k2a = r02y, k2b = r13y; // uniform key

  __shared__ float bv[128];
  __shared__ int   bi[128];
  __shared__ int   take_s;

  float best = -INFINITY; int besti = 0;
  const int half = (B_SZ * VOCD) / 2;
  for (int v = tid; v < VOCD; v += 128) {
    int j = b * VOCD + v;
    uint32_t o0, o1, bits;
    if (j < half) { tf2x32(k1a, k1b, (uint32_t)j, (uint32_t)(j + half), o0, o1); bits = o0; }
    else          { tf2x32(k1a, k1b, (uint32_t)(j - half), (uint32_t)j, o0, o1); bits = o1; }
    float u = fmaxf(bits_to_u01(bits), 1.1754944e-38f);
    float g = -__logf(-__logf(u));
    float s = pyp[(size_t)b * VOCD + v] + g;
    if (s > best) { best = s; besti = v; }
  }
  bv[tid] = best; bi[tid] = besti;
  __syncthreads();
  for (int s = 64; s > 0; s >>= 1) {
    if (tid < s) {
      if (bv[tid + s] > bv[tid] || (bv[tid + s] == bv[tid] && bi[tid + s] < bi[tid])) {
        bv[tid] = bv[tid + s]; bi[tid] = bi[tid + s];
      }
    }
    __syncthreads();
  }
  if (tid == 0) {
    uint32_t o0, o1, bits;
    if (b < B_SZ / 2) { tf2x32(k2a, k2b, (uint32_t)b, (uint32_t)(b + B_SZ / 2), o0, o1); bits = o0; }
    else              { tf2x32(k2a, k2b, (uint32_t)(b - B_SZ / 2), (uint32_t)b, o0, o1); bits = o1; }
    take_s = (blend[0] != 0) && (bits_to_u01(bits) < 0.1f);
  }
  __syncthreads();
  const int take = take_s, samp = bi[0];
  float* row = rnni + (size_t)b * RNNI_LD;
  for (int i = tid; i < DECD; i += 128) row[i] = psv[(size_t)b * DECD + i];
  for (int v = tid; v < VOCD; v += 128)
    row[DECD + v] = take ? (v == samp ? 1.0f : 0.0f) : yin[(size_t)b * VOCD + v];
  for (int i = tid; i < 2 * LISD; i += 128)
    row[DECD + VOCD + i] = pcv[(size_t)b * 2 * LISD + i];
  if (tid < RNNI_LD - DIN1) row[DIN1 + tid] = 0.0f; // zero the pad columns
}

// ---------------------------------------------------------------------------
// Kernel 2: C[M,N] = act(A1[M,K1]@B1[K1,N] + A2[M,K2]@B2[K2,N] + bias)
// bf16 WMMA, f32 accumulate. 128x128 block tile, 8 waves of 32x64.
// Double-buffered LDS + register prefetch; vectorized float4 staging.
// Requires: lda* multiple of 4 and lda* >= round_up(K*,4); rows 16B-aligned.
// ---------------------------------------------------------------------------
__global__ __launch_bounds__(256)
void gemm2_wmma_bf16(const float* __restrict__ A1, int lda1,
                     const float* __restrict__ B1, int K1,
                     const float* __restrict__ A2, int lda2,
                     const float* __restrict__ B2, int K2,
                     const float* __restrict__ bias, float* __restrict__ C,
                     int N, int act) {
  __shared__ __align__(16) __bf16 As[2][128][LDSP]; // [m][k]
  __shared__ __align__(16) __bf16 Bs[2][128][LDSP]; // [n][k] (transposed)

  const int tid  = threadIdx.x;
  const int lane = tid & 31, wave = tid >> 5;
  const int m0 = blockIdx.y * 128, n0 = blockIdx.x * 128;
  const int wm = (wave & 3) * 32, wn = (wave >> 2) * 64;
  const int ml   = lane & 15;
  const int aoff = (lane >= 16) ? 8 : 0;   // ISA 16-bit A-matrix K base per lane half
  const int boff = (lane >= 16) ? 8 : 0;   // B-matrix K base per lane half (matches A)

  const int ns1 = (K1 + 31) >> 5;
  const int ns2 = (K2 + 31) >> 5;
  const int total = ns1 + ns2;

  float4 aR[4];     // staged A: 4 x float4 per thread (128x32 tile)
  float4 bR[2][2];  // staged B: 2 row-pair x 2 rows of float4 (32x128 tile)

  auto load_regs = [&](int s) {
    const float* A; const float* Bm; int lda, Kb, k0;
    if (s < ns1) { A = A1; lda = lda1; Bm = B1; Kb = K1; k0 = s << 5; }
    else         { A = A2; lda = lda2; Bm = B2; Kb = K2; k0 = (s - ns1) << 5; }
#pragma unroll
    for (int j = 0; j < 4; ++j) {
      int e = tid + 256 * j;
      int r = e >> 3, c4 = (e & 7) << 2;
      int gk = k0 + c4;
      float4 v = make_float4(0.f, 0.f, 0.f, 0.f);
      if (gk < Kb) v = *(const float4*)(A + (size_t)(m0 + r) * lda + gk);
      aR[j] = v;
    }
#pragma unroll
    for (int j = 0; j < 2; ++j) {
      int e = tid + 256 * j;
      int rp = e >> 5, c4 = (e & 31) << 2;
      int gn = n0 + c4;
#pragma unroll
      for (int h = 0; h < 2; ++h) {
        int gk = k0 + 2 * rp + h;
        float4 v = make_float4(0.f, 0.f, 0.f, 0.f);
        if (gk < Kb) {
          const float* p = Bm + (size_t)gk * N + gn;
          if (((N & 3) == 0) && (gn + 4 <= N)) {
            v = *(const float4*)p;
          } else {
            if (gn < N)     v.x = p[0];
            if (gn + 1 < N) v.y = p[1];
            if (gn + 2 < N) v.z = p[2];
            if (gn + 3 < N) v.w = p[3];
          }
        }
        bR[j][h] = v;
      }
    }
  };

  auto store_regs = [&](int buf) {
#pragma unroll
    for (int j = 0; j < 4; ++j) {
      int e = tid + 256 * j;
      int r = e >> 3, c4 = (e & 7) << 2;
      union { __bf16 h[4]; uint2 u; } t;
      t.h[0] = (__bf16)aR[j].x; t.h[1] = (__bf16)aR[j].y;
      t.h[2] = (__bf16)aR[j].z; t.h[3] = (__bf16)aR[j].w;
      *(uint2*)&As[buf][r][c4] = t.u; // ds_store_b64
    }
#pragma unroll
    for (int j = 0; j < 2; ++j) {
      int e = tid + 256 * j;
      int rp = e >> 5, c4 = (e & 31) << 2;
#pragma unroll
      for (int q = 0; q < 4; ++q) {
        union { __bf16 h[2]; uint32_t u; } p;
        p.h[0] = (__bf16)((&bR[j][0].x)[q]); // row 2*rp   (even K)
        p.h[1] = (__bf16)((&bR[j][1].x)[q]); // row 2*rp+1 (odd K)
        *(uint32_t*)&Bs[buf][c4 + q][2 * rp] = p.u; // ds_store_b32, K-pair packed
      }
    }
  };

  v8f acc[2][4];
#pragma unroll
  for (int i = 0; i < 2; ++i)
#pragma unroll
    for (int j = 0; j < 4; ++j)
#pragma unroll
      for (int e = 0; e < 8; ++e) acc[i][j][e] = 0.0f;

  load_regs(0);
  store_regs(0);
  int cur = 0;
  for (int s = 0; s < total; ++s) {
    const bool has_next = (s + 1 < total);
    if (has_next) load_regs(s + 1);   // global prefetch for next tile
    __syncthreads();                   // LDS[cur] ready for all waves

    Frag a[2], b[4];
#pragma unroll
    for (int i = 0; i < 2; ++i) {
      const __bf16* p = &As[cur][wm + 16 * i + ml][0];
      a[i].q[0] = *(const int4*)(p + aoff);       // K 0..7   (or 8..15)
      a[i].q[1] = *(const int4*)(p + aoff + 16);  // K 16..23 (or 24..31)
    }
#pragma unroll
    for (int j = 0; j < 4; ++j) {
      const __bf16* p = &Bs[cur][wn + 16 * j + ml][0];
      b[j].q[0] = *(const int4*)(p + boff);       // K 0..7   (or 8..15)
      b[j].q[1] = *(const int4*)(p + boff + 16);  // K 16..23 (or 24..31)
    }
#pragma unroll
    for (int i = 0; i < 2; ++i)
#pragma unroll
      for (int j = 0; j < 4; ++j)
        acc[i][j] = __builtin_amdgcn_wmma_f32_16x16x32_bf16(
            false, a[i].v, false, b[j].v, (short)0, acc[i][j], false, false);

    if (has_next) { store_regs(cur ^ 1); cur ^= 1; }
  }

  // Epilogue: 16x16 f32 C layout (lane<16: M=r, N=lane; lane>=16: M=8+r, N=lane-16)
  const int mb = (lane >= 16) ? 8 : 0;
#pragma unroll
  for (int i = 0; i < 2; ++i)
#pragma unroll
    for (int j = 0; j < 4; ++j) {
      int n = n0 + wn + 16 * j + ml;
      if (n < N) {
        float bvv = bias[n];
#pragma unroll
        for (int r = 0; r < 8; ++r) {
          int m = m0 + wm + 16 * i + mb + r;
          float v = acc[i][j][r] + bvv;
          if (act) v = fmaxf(v, 0.0f);
          C[(size_t)m * N + n] = v;
        }
      }
    }
}

// ---------------------------------------------------------------------------
// Kernel 3: LSTM cell pointwise (gates i,f,g,o)
// ---------------------------------------------------------------------------
__global__ __launch_bounds__(256)
void lstm_pointwise_kernel(const float* __restrict__ z, const float* __restrict__ c_in,
                           float* __restrict__ h_out, float* __restrict__ c_out) {
  int idx = blockIdx.x * 256 + threadIdx.x;
  if (idx >= B_SZ * DECD) return;
  int b = idx >> 9, j = idx & 511;
  const float* zr = z + (size_t)b * 4 * DECD;
  float gi = zr[j], gf = zr[DECD + j], gg = zr[2 * DECD + j], go = zr[3 * DECD + j];
  float si = 1.0f / (1.0f + __expf(-gi));
  float sf = 1.0f / (1.0f + __expf(-gf));
  float so = 1.0f / (1.0f + __expf(-go));
  float cn = sf * c_in[idx] + si * tanhf(gg);
  c_out[idx] = cn;
  h_out[idx] = so * tanhf(cn);
}

// ---------------------------------------------------------------------------
// Kernel 4: per-row attention: scores -> softmax -> context.
// Streams ~2MB per block exactly once -> non-temporal loads so the 1GB of
// keys/features doesn't evict resident weights from the 192MB L2.
// ---------------------------------------------------------------------------
__global__ __launch_bounds__(256)
void attention_kernel(const float* __restrict__ q, const float* __restrict__ keys,
                      const unsigned char* __restrict__ mask,
                      const float* __restrict__ feats, float* __restrict__ ci) {
  const int b = blockIdx.x, tid = threadIdx.x;
  __shared__ float qs[ATTD];
  __shared__ float sc[T_SZ];
  __shared__ float red[256];

  for (int i = tid; i < ATTD; i += 256) qs[i] = q[(size_t)b * ATTD + i];
  __syncthreads();

  for (int t = tid; t < T_SZ; t += 256) {
    const f32x4* k4 = (const f32x4*)(keys + ((size_t)b * T_SZ + t) * ATTD);
    const f32x4* q4 = (const f32x4*)qs;
    float acc = 0.0f;
#pragma unroll 8
    for (int i = 0; i < ATTD / 4; ++i) {
      f32x4 a = __builtin_nontemporal_load(k4 + i); // NT: single-use stream
      f32x4 c = q4[i];
      acc += a.x * c.x + a.y * c.y + a.z * c.z + a.w * c.w;
    }
    unsigned char mk = __builtin_nontemporal_load(mask + (size_t)b * T_SZ + t);
    sc[t] = mk ? acc : -1e9f;
  }
  __syncthreads();

  float m = -INFINITY;
  for (int t = tid; t < T_SZ; t += 256) m = fmaxf(m, sc[t]);
  red[tid] = m; __syncthreads();
  for (int s = 128; s > 0; s >>= 1) {
    if (tid < s) red[tid] = fmaxf(red[tid], red[tid + s]);
    __syncthreads();
  }
  m = red[0]; __syncthreads();

  float sum = 0.0f;
  for (int t = tid; t < T_SZ; t += 256) { float e = __expf(sc[t] - m); sc[t] = e; sum += e; }
  red[tid] = sum; __syncthreads();
  for (int s = 128; s > 0; s >>= 1) {
    if (tid < s) red[tid] += red[tid + s];
    __syncthreads();
  }
  const float inv = 1.0f / red[0];
  __syncthreads();
  for (int t = tid; t < T_SZ; t += 256) sc[t] *= inv;
  __syncthreads();

  for (int d = tid; d < 2 * LISD; d += 256) {
    float acc = 0.0f;
    for (int t = 0; t < T_SZ; ++t)
      acc += sc[t] * __builtin_nontemporal_load(feats + ((size_t)b * T_SZ + t) * (2 * LISD) + d);
    ci[(size_t)b * (2 * LISD) + d] = acc;
  }
}

// ---------------------------------------------------------------------------
extern "C" void kernel_launch(void* const* d_in, const int* in_sizes, int n_in,
                              void* d_out, int out_size, void* d_ws, size_t ws_size,
                              hipStream_t stream) {
  const float* yin   = (const float*)d_in[0];
  const float* h1    = (const float*)d_in[1];
  const float* c1    = (const float*)d_in[2];
  const float* h2    = (const float*)d_in[3];
  const float* c2    = (const float*)d_in[4];
  const float* psv   = (const float*)d_in[5];
  const float* pcv   = (const float*)d_in[6];
  const float* pyp   = (const float*)d_in[7];
  const float* feats = (const float*)d_in[8];
  const float* keys  = (const float*)d_in[9];
  const unsigned char* mask = (const unsigned char*)d_in[10];
  const int*   blend = (const int*)d_in[11];
  const float* W1k = (const float*)d_in[12];
  const float* W1r = (const float*)d_in[13];
  const float* b1  = (const float*)d_in[14];
  const float* W2k = (const float*)d_in[15];
  const float* W2r = (const float*)d_in[16];
  const float* b2  = (const float*)d_in[17];
  const float* Wp1 = (const float*)d_in[18];
  const float* bp1 = (const float*)d_in[19];
  const float* Wp2 = (const float*)d_in[20];
  const float* bp2 = (const float*)d_in[21];
  const float* Wc1 = (const float*)d_in[22];
  const float* bc1 = (const float*)d_in[23];
  const float* Wc2 = (const float*)d_in[24];
  const float* bc2 = (const float*)d_in[25];

  // Outputs concatenated: yp, o1, c1n, si, c2n, ci
  float* out = (float*)d_out;
  float* yp  = out;
  float* o1  = yp  + (size_t)B_SZ * VOCD;
  float* c1n = o1  + (size_t)B_SZ * DECD;
  float* si  = c1n + (size_t)B_SZ * DECD;
  float* c2n = si  + (size_t)B_SZ * DECD;
  float* ci  = c2n + (size_t)B_SZ * DECD;

  // Workspace (aliased regions; ~6.6 MB peak)
  float* ws   = (float*)d_ws;
  float* rnni = ws;                                  // 512*1152, later reused by t1/t2
  float* z    = rnni + (size_t)B_SZ * RNNI_LD;       // 512*2048, later reused by q
  float* t1   = rnni;                                // 512*1024 (rnni dead after LSTM1 GEMM)
  float* q    = z;                                   // 512*512  (z dead after 2nd pointwise)
  float* t2   = rnni;                                // 512*492  (t1 dead after q GEMM)

  dim3 blk(256);

  // 1) scheduled sampling + concat (rnni padded to ld 1152)
  sample_concat_kernel<<<B_SZ, 128, 0, stream>>>(yin, pyp, psv, pcv, blend, rnni);
  // 2) LSTM1: z = rnni@W1k + h1@W1r + b1   [512,2048]
  gemm2_wmma_bf16<<<dim3(16, 4), blk, 0, stream>>>(rnni, RNNI_LD, W1k, DIN1,
                                                   h1, DECD, W1r, DECD, b1, z, 4 * DECD, 0);
  lstm_pointwise_kernel<<<(B_SZ * DECD) / 256, blk, 0, stream>>>(z, c1, o1, c1n);
  // 3) LSTM2: z = o1@W2k + h2@W2r + b2
  gemm2_wmma_bf16<<<dim3(16, 4), blk, 0, stream>>>(o1, DECD, W2k, DECD,
                                                   h2, DECD, W2r, DECD, b2, z, 4 * DECD, 0);
  lstm_pointwise_kernel<<<(B_SZ * DECD) / 256, blk, 0, stream>>>(z, c2, si, c2n);
  // 4) query MLP: t1 = relu(si@Wp1+bp1) [512,1024]; q = t1@Wp2+bp2 [512,512]
  gemm2_wmma_bf16<<<dim3(8, 4), blk, 0, stream>>>(si, DECD, Wp1, DECD,
                                                  nullptr, 0, nullptr, 0, bp1, t1, 2 * ATTD, 1);
  gemm2_wmma_bf16<<<dim3(4, 4), blk, 0, stream>>>(t1, 2 * ATTD, Wp2, 2 * ATTD,
                                                  nullptr, 0, nullptr, 0, bp2, q, ATTD, 0);
  // 5) attention (HBM-streaming dominant stage: ~1 GB -> ~46us floor at 23.3 TB/s)
  attention_kernel<<<B_SZ, blk, 0, stream>>>(q, keys, mask, feats, ci);
  // 6) char MLP: t2 = relu(si@Wc1_top + ci@Wc1_bot + bc1) [512,492]; yp = t2@Wc2+bc2 [512,123]
  gemm2_wmma_bf16<<<dim3(4, 4), blk, 0, stream>>>(si, DECD, Wc1, DECD,
                                                  ci, 2 * LISD, Wc1 + (size_t)DECD * (4 * VOCD), 2 * LISD,
                                                  bc1, t2, 4 * VOCD, 1);
  gemm2_wmma_bf16<<<dim3(1, 4), blk, 0, stream>>>(t2, 4 * VOCD, Wc2, 4 * VOCD,
                                                  nullptr, 0, nullptr, 0, bc2, yp, VOCD, 0);
}
